// DHT_16527034155157
// MI455X (gfx1250) — matrix operand order repaired and checked by reading, not compiled, not run on hardware
//
#include <hip/hip_runtime.h>
#include <hip/hip_bf16.h>

// ---------------------------------------------------------------------------
// Deep Hough Transform on MI455X (gfx1250)
//   out[bc][a][r] = sum_{pixels p : rho_bin(a,p)==r} x[bc][p]
//
// Per (angle, 16-rho-bin window) block the segment-sum becomes WMMA:
//   D[16 bins x 16 ch] += OneHot[16 bins x 32 pix] * X[32 pix x 16 ch]
// via v_wmma_f32_16x16x32_f16 with an f16 hi/lo split of x (~fp32 accuracy).
// Along the dominant axis rho is monotone (|slope| >= 0.695), so all pixels of
// a line mapping into a 16-bin window fit one 8-aligned 32-pixel K-window.
// No atomics: every output element is produced by exactly one wave.
// ---------------------------------------------------------------------------

typedef __attribute__((ext_vector_type(16))) _Float16 v16h;
typedef __attribute__((ext_vector_type(8)))  _Float16 v8h;
typedef __attribute__((ext_vector_type(8)))  float    v8f;
typedef __attribute__((ext_vector_type(4)))  float    v4f;

#define NUM_ANGLE 180
#define NUM_RHO   180
#define HH 128
#define WW 128
#define HW 16384
#define BC 1024

// ---------------------------------------------------------------------------
// Stage 1: rho-bin tables (double precision, ties-to-even like np.round),
// row-major (h*128+w) and transposed (w*128+h), plus float cos/sin tables.
// ---------------------------------------------------------------------------
__global__ __launch_bounds__(256)
void dht_tables(unsigned char* __restrict__ ridx, unsigned char* __restrict__ ridxT,
                float* __restrict__ tabc, float* __restrict__ tabs)
{
  const int a = blockIdx.x;
  const double itheta = 3.14159265358979323846 / (double)NUM_ANGLE;
  const double irho   = 182.0 / 179.0;   // int(sqrt(128^2+128^2)+1) / (numRho-1)
  const double ang = (double)a * itheta;
  const double tc = cos(ang) / irho;
  const double ts = sin(ang) / irho;
  if (threadIdx.x == 0) { tabc[a] = (float)tc; tabs[a] = (float)ts; }
  for (int i = threadIdx.x; i < HW; i += 256) {
    const int h = i >> 7, w = i & 127;
    const double xc = (double)(w - 64), yc = (double)(h - 64);
    // numpy: products rounded to double, then added (no FMA), round half-even
    const double val = __dadd_rn(__dmul_rn(xc, tc), __dmul_rn(yc, ts));
    int r = (int)nearbyint(val) + NUM_RHO / 2;
    r = r < 0 ? 0 : (r > NUM_RHO - 1 ? NUM_RHO - 1 : r);
    ridx [a * HW + i]            = (unsigned char)r;
    ridxT[a * HW + (w << 7) + h] = (unsigned char)r;
  }
}

// ---------------------------------------------------------------------------
// Stage 2: split x into f16 hi + f16 residual planes, native and transposed
// (transpose through LDS so both sides stay coalesced). x = hi + lo.
// ---------------------------------------------------------------------------
__global__ __launch_bounds__(256)
void dht_split(const float* __restrict__ x,
               _Float16* __restrict__ xh,  _Float16* __restrict__ xl,
               _Float16* __restrict__ xhT, _Float16* __restrict__ xlT)
{
  __shared__ _Float16 th[32][130];   // +2 pad: conflict-free transposed reads
  __shared__ _Float16 tl[32][130];
  const int bc   = blockIdx.x;
  const int h0   = blockIdx.y * 32;
  const int t    = threadIdx.x;
  const int base = bc * HW;
#pragma unroll
  for (int k = 0; k < 16; ++k) {
    const int i  = t + k * 256;          // 0..4095 within 32x128 tile
    const int lh = i >> 7, w = i & 127;
    const float    v  = x[base + (h0 + lh) * WW + w];
    const _Float16 hi = (_Float16)v;
    const _Float16 lo = (_Float16)(v - (float)hi);
    xh[base + (h0 + lh) * WW + w] = hi;
    xl[base + (h0 + lh) * WW + w] = lo;
    th[lh][w] = hi;
    tl[lh][w] = lo;
  }
  __syncthreads();
#pragma unroll
  for (int k = 0; k < 16; ++k) {
    const int o  = t + k * 256;
    const int ow = o >> 5, oh = o & 31;
    xhT[base + ow * HH + h0 + oh] = th[oh][ow];
    xlT[base + ow * HH + h0 + oh] = tl[oh][ow];
  }
}

// ---------------------------------------------------------------------------
// Stage 3: WMMA segment-sum.
//   grid = (180 angles, 12 rho-windows), block = 512 threads = 16 waves.
//   One barrier per processed line (double-buffered one-hot A in LDS);
//   empty lines (window outside the image) are skipped block-uniformly.
//   B loads use one lane-invariant base + constant IOFFSETs per channel tile.
// ---------------------------------------------------------------------------
__global__ __launch_bounds__(512)
void dht_wmma(const _Float16* __restrict__ xh,  const _Float16* __restrict__ xl,
              const _Float16* __restrict__ xhT, const _Float16* __restrict__ xlT,
              const unsigned char* __restrict__ ridx,
              const unsigned char* __restrict__ ridxT,
              const float* __restrict__ tabc, const float* __restrict__ tabs,
              float* __restrict__ out)
{
  __shared__ alignas(32) _Float16 ldsA[2][512];  // double-buffered one-hot A

  const int a  = blockIdx.x;          // angle
  const int r0 = blockIdx.y << 4;     // rho window [r0, r0+16)
  const int t  = threadIdx.x;
  const int L  = t & 31;              // lane within wave32
  const int wv = t >> 5;              // wave id 0..15

  const float cN = tabc[a];
  const float sN = tabs[a];
  const bool  rowmaj = fabsf(cN) >= fabsf(sN);          // traverse along w else h
  const unsigned char* __restrict__ pr = (rowmaj ? ridx : ridxT) + a * HW;
  const float su = rowmaj ? cN : sN;  // rho slope along traversal axis, |su|>=0.695
  const float sv = rowmaj ? sN : cN;
  const float inv_su = 1.0f / su;
  const float c0 = (float)(r0 - 90) - 0.5f;

  // thread t fills ldsA[b][t] = A[m][k] in this kernel's contiguous-K A-operand order (B matches):
  // lanes 0-15 hold K {0..15}, lanes 16-31 hold K {16..31}.
  const int bL = t >> 4;                                   // A-operand lane
  const int bi = t & 15;                                   // half index in lane
  const int bk = bi + ((bL & 16) ? 16 : 0);
  const int bm = bL & 15;                                  // rho row of A

  // B operand: lane n (0-15) = channel, K 0..15; lane n+16 = same chans, K 16..31.
  const int chv   = wv * 64 + (L & 15);       // channel (q*16 added via IOFFSET)
  const int khalf = (L >> 4) * 16;
  const _Float16* __restrict__ phl = (rowmaj ? xh : xhT) + chv * HW + khalf;
  const _Float16* __restrict__ pll = (rowmaj ? xl : xlT) + chv * HW + khalf;

  v8f acc[4] = {v8f{}, v8f{}, v8f{}, v8f{}};
  int pb = 0;

  for (int v = 0; v < 128; ++v) {
    // Window start u0: every pixel on this line with bin in [r0,r0+16) lies in
    // [u0, u0+32). Run <= 16/0.695 = 23.0 px; floor + 8-align costs <= 8, so
    // 23 + 8 + fuzz < 32. 8-alignment keeps the B loads 16-byte aligned.
    const float K  = (float)(v - 64) * sv - 64.0f * su;
    const float e1 = (c0 - K) * inv_su;
    const float e2 = e1 + 16.0f * inv_su;
    const float emin = fminf(e1, e2);
    const float emax = fmaxf(e1, e2);
    if (emax < -1.0f || emin > 128.0f) continue;   // uniform: window off-image

    int u0 = (int)floorf(emin);
    u0 &= ~7;
    u0 = u0 < 0 ? 0 : (u0 > 96 ? 96 : u0);
    const int lineoff = (v << 7) + u0;             // uniform per line

    // one-hot A from the rho table (32 bytes, L0-resident; 16x redundant reads)
    ldsA[pb][t] = (_Float16)((((int)pr[lineoff + bk] - r0) == bm) ? 1.0f : 0.0f);
    if (t == 0 && v + 1 < 128)                     // gfx1250 global_prefetch path
      __builtin_prefetch(pr + lineoff + 128, 0, 3);
    __syncthreads();                               // A visible; prev-buf reads done

    const v16h A = *(const v16h*)(&ldsA[pb][L * 16]);   // 32B per lane, aligned
    const _Float16* __restrict__ pH = phl + lineoff;
    const _Float16* __restrict__ pL = pll + lineoff;
    if (v + 1 < 128) __builtin_prefetch(pH + 128, 0, 3);

#pragma unroll
    for (int q = 0; q < 4; ++q) {
      union { v16h v; v8h h[2]; } Bh, Bl;
      Bh.h[0] = *(const v8h*)(pH + q * 16 * HW);        // q-offsets fold into
      Bh.h[1] = *(const v8h*)(pH + q * 16 * HW + 8);    // the 24-bit IOFFSET
      Bl.h[0] = *(const v8h*)(pL + q * 16 * HW);
      Bl.h[1] = *(const v8h*)(pL + q * 16 * HW + 8);
      acc[q] = __builtin_amdgcn_wmma_f32_16x16x32_f16(
                   false, A, false, Bh.v, (short)0, acc[q], false, false);
      acc[q] = __builtin_amdgcn_wmma_f32_16x16x32_f16(
                   false, A, false, Bl.v, (short)0, acc[q], false, false);
    }
    pb ^= 1;
  }

  // D layout: lanes 0-15 -> rows M=0..7 in VGPR 0..7, lanes 16-31 -> M=8..15.
  // Per lane: 8 consecutive rho values from r0+mbase -> two 16B stores.
  const int mbase = (L >> 4) * 8;
#pragma unroll
  for (int q = 0; q < 4; ++q) {
    const int ch = chv + q * 16;
    const int ob = (ch * NUM_ANGLE + a) * NUM_RHO + r0 + mbase;
    const v8f d = acc[q];
    if (r0 + mbase + 3 < NUM_RHO) {                // groups are 4-aligned in r
      v4f s0 = {d[0], d[1], d[2], d[3]};
      *(v4f*)(out + ob) = s0;
    }
    if (r0 + mbase + 7 < NUM_RHO) {
      v4f s1 = {d[4], d[5], d[6], d[7]};
      *(v4f*)(out + ob + 4) = s1;
    }
  }
}

// ---------------------------------------------------------------------------
// Launch: tables -> split -> wmma, all on `stream`.
// Workspace: 2x2.95MB rho tables + 4x32MB f16 planes ~= 134 MB.
// ---------------------------------------------------------------------------
extern "C" void kernel_launch(void* const* d_in, const int* in_sizes, int n_in,
                              void* d_out, int out_size, void* d_ws, size_t ws_size,
                              hipStream_t stream) {
  (void)in_sizes; (void)n_in; (void)out_size; (void)ws_size;
  const float* x = (const float*)d_in[0];
  float* out = (float*)d_out;

  char* ws = (char*)d_ws;
  size_t off = 0;
  auto take = [&](size_t bytes) -> void* {
    void* p = ws + off;
    off = (off + bytes + 255) & ~(size_t)255;
    return p;
  };
  unsigned char* ridx  = (unsigned char*)take((size_t)NUM_ANGLE * HW);
  unsigned char* ridxT = (unsigned char*)take((size_t)NUM_ANGLE * HW);
  float* tabc = (float*)take(NUM_ANGLE * sizeof(float));
  float* tabs = (float*)take(NUM_ANGLE * sizeof(float));
  _Float16* xh  = (_Float16*)take((size_t)BC * HW * 2);
  _Float16* xl  = (_Float16*)take((size_t)BC * HW * 2);
  _Float16* xhT = (_Float16*)take((size_t)BC * HW * 2);
  _Float16* xlT = (_Float16*)take((size_t)BC * HW * 2);

  dht_tables<<<dim3(NUM_ANGLE), dim3(256), 0, stream>>>(ridx, ridxT, tabc, tabs);
  dht_split<<<dim3(BC, 4), dim3(256), 0, stream>>>(x, xh, xl, xhT, xlT);
  dht_wmma<<<dim3(NUM_ANGLE, 12), dim3(512), 0, stream>>>(
      xh, xl, xhT, xlT, ridx, ridxT, tabc, tabs, out);
}
